// LIMIXChannelAttention_73564199846411
// MI455X (gfx1250) — compile-verified, matrix operand order checked by reading, not run on hardware
//
#include <hip/hip_runtime.h>

typedef _Float16 h8   __attribute__((ext_vector_type(8)));
typedef _Float16 v16h __attribute__((ext_vector_type(16)));
typedef float    v8f  __attribute__((ext_vector_type(8)));

#define D_EMB   768
#define C_SLOTS 6
#define SLOT    128
#define T_TOK   8                    // tokens per block
#define ROWS    (T_TOK * C_SLOTS)    // 48 GEMM rows per block
#define PADW    136                  // 128 + 8 halves pad: 272B row stride, conflict-free b128

// ---------------------------------------------------------------------------
// Weight conversion: fp32 -> f16, laid out so each WMMA B-matrix column is
// K-contiguous. qkv_proj (3,2,64,128) flattens to exactly [n=0..383][k=0..127]
// (pure convert). out_proj (2,64,128) needs transpose: wout[s][hd].
// ---------------------------------------------------------------------------
__global__ void convert_weights(const float* __restrict__ qkvp,
                                const float* __restrict__ outp,
                                _Float16* __restrict__ wqkv,
                                _Float16* __restrict__ wout) {
  int i = blockIdx.x * 256 + threadIdx.x;
  if (i < 3 * 2 * 64 * 128) wqkv[i] = (_Float16)qkvp[i];
  if (i < 128 * 128) {
    int hd = i & 127, s = i >> 7;
    wout[s * 128 + hd] = (_Float16)outp[hd * 128 + s];
  }
}

// Load a 16-half WMMA fragment: two contiguous 8-half (16B) runs at K and K+16.
__device__ __forceinline__ v16h load_frag(const _Float16* p) {
  union { v16h v; h8 h[2]; } u;
  u.h[0] = *(const h8*)p;
  u.h[1] = *(const h8*)(p + 16);
  return u.v;
}

__global__ void __launch_bounds__(256)
attn_main(const float* __restrict__ x, const float* __restrict__ lnw,
          const float* __restrict__ lnb, const _Float16* __restrict__ wqkv,
          const _Float16* __restrict__ wout, float* __restrict__ out) {
  __shared__ __align__(16) _Float16 sXN[ROWS * PADW];        // LN out, reused for attn out
  __shared__ __align__(16) _Float16 sQKV[3 * ROWS * PADW];   // [j][r = t*6+c][h*64 + d]

  const int tid  = threadIdx.x;
  const int lane = tid & 31;
  const int wave = tid >> 5;
  const long rowBase = (long)blockIdx.x * (T_TOK * D_EMB);   // flat float offset of block

  // ---- Stage 1: LayerNorm, one token per wave, f16 rows into sXN ----------
  {
    const int t = wave;
    const float* xp = x + rowBase + t * D_EMB;
    float vals[24];
    float s = 0.f, s2 = 0.f;
#pragma unroll
    for (int k = 0; k < 24; ++k) {
      float v = xp[lane + 32 * k];
      vals[k] = v; s += v; s2 += v * v;
    }
#pragma unroll
    for (int off = 16; off > 0; off >>= 1) {
      s  += __shfl_xor(s,  off, 32);
      s2 += __shfl_xor(s2, off, 32);
    }
    float mean = s * (1.f / 768.f);
    float var  = s2 * (1.f / 768.f) - mean * mean;
    float rstd = rsqrtf(var + 1e-5f);
#pragma unroll
    for (int k = 0; k < 24; ++k) {
      int idx = lane + 32 * k;
      float xv = (vals[k] - mean) * rstd * lnw[idx] + lnb[idx];
      // row r = t*6 + idx/128, col idx%128  (contiguous 128-blocks per row)
      sXN[(t * C_SLOTS + (idx >> 7)) * PADW + (idx & 127)] = (_Float16)xv;
    }
  }
  __syncthreads();

  // ---- Stage 2: QKV GEMM (48x128)@(128x384), f16 WMMA, fp32 accum ---------
  {
    const int m16 = lane & 15, hk = lane >> 4;
#pragma unroll 1
    for (int i = 0; i < 9; ++i) {               // 3 mtiles * 24 ntiles / 8 waves
      int job = wave + 8 * i;
      int mt = job / 24, nt = job % 24;
      v8f acc = {};
#pragma unroll
      for (int ks = 0; ks < 4; ++ks) {
        v16h a = load_frag(&sXN[(mt * 16 + m16) * PADW + ks * 32 + hk * 8]);
        v16h b = load_frag(&wqkv[(nt * 16 + m16) * SLOT + ks * 32 + hk * 8]);
        acc = __builtin_amdgcn_wmma_f32_16x16x32_f16(false, a, false, b,
                                                     (short)0, acc, false, false);
      }
      int j = nt >> 3;                           // 0:q 1:k 2:v
      int nBase = (nt & 7) * 16 + m16;           // hd column 0..127 within q/k/v
      _Float16* dst = &sQKV[(j * ROWS + mt * 16 + hk * 8) * PADW + nBase];
#pragma unroll
      for (int v = 0; v < 8; ++v)
        dst[v * PADW] = (_Float16)acc[v];        // row r = mt*16 + hk*8 + v
    }
  }
  __syncthreads();

  // ---- Stage 3: per (token, head, channel) attention, C=6, d=64 -----------
  if (tid < T_TOK * 2 * C_SLOTS) {               // 96 threads
    int t = tid / (2 * C_SLOTS);
    int rem = tid % (2 * C_SLOTS);
    int h = rem / C_SLOTS, c = rem % C_SLOTS;
    const _Float16* q  = &sQKV[(0 * ROWS + t * C_SLOTS + c) * PADW + h * 64];
    const _Float16* kk = &sQKV[(1 * ROWS + t * C_SLOTS) * PADW + h * 64];
    const _Float16* vv = &sQKV[(2 * ROWS + t * C_SLOTS) * PADW + h * 64];
    float p[6];
    float mx = -1e30f;
#pragma unroll
    for (int e = 0; e < 6; ++e) {
      float sdot = 0.f;
      for (int d2 = 0; d2 < 64; ++d2)
        sdot += (float)q[d2] * (float)kk[e * PADW + d2];
      p[e] = sdot * 0.125f;                      // SCALE = 64^-0.5
      mx = fmaxf(mx, p[e]);
    }
    float den = 0.f;
#pragma unroll
    for (int e = 0; e < 6; ++e) { p[e] = __expf(p[e] - mx); den += p[e]; }
    float rden = 1.f / den;
#pragma unroll
    for (int e = 0; e < 6; ++e) p[e] *= rden;
    _Float16* orow = &sXN[(t * C_SLOTS + c) * PADW + h * 64];
    for (int d2 = 0; d2 < 64; ++d2) {
      float o = 0.f;
#pragma unroll
      for (int e = 0; e < 6; ++e) o += p[e] * (float)vv[e * PADW + d2];
      orow[d2] = (_Float16)o;
    }
  }
  __syncthreads();

  // ---- Stage 4: out projection (48x128)@(128x128) + residual --------------
  // Global address needs no (t,c) split: t*768 + c*128 == r*128.
  {
    const int m16 = lane & 15, hk = lane >> 4;
#pragma unroll 1
    for (int i = 0; i < 3; ++i) {               // 3 mtiles * 8 ntiles / 8 waves
      int job = wave + 8 * i;
      int mt = job >> 3, nt = job & 7;
      v8f acc = {};
#pragma unroll
      for (int ks = 0; ks < 4; ++ks) {
        v16h a = load_frag(&sXN[(mt * 16 + m16) * PADW + ks * 32 + hk * 8]);
        v16h b = load_frag(&wout[(nt * 16 + m16) * SLOT + ks * 32 + hk * 8]);
        acc = __builtin_amdgcn_wmma_f32_16x16x32_f16(false, a, false, b,
                                                     (short)0, acc, false, false);
      }
      long base = rowBase + (long)(mt * 16 + hk * 8) * SLOT + nt * 16 + m16;
#pragma unroll
      for (int v = 0; v < 8; ++v) {
        long addr = base + (long)v * SLOT;
        out[addr] = x[addr] + acc[v];
      }
    }
  }
}

extern "C" void kernel_launch(void* const* d_in, const int* in_sizes, int n_in,
                              void* d_out, int out_size, void* d_ws, size_t ws_size,
                              hipStream_t stream) {
  const float* x    = (const float*)d_in[0];
  const float* qkvp = (const float*)d_in[1];
  const float* outp = (const float*)d_in[2];
  const float* lnw  = (const float*)d_in[3];
  const float* lnb  = (const float*)d_in[4];
  float* out = (float*)d_out;

  _Float16* wqkv = (_Float16*)d_ws;        // 49152 halves
  _Float16* wout = wqkv + 49152;           // 16384 halves

  convert_weights<<<192, 256, 0, stream>>>(qkvp, outp, wqkv, wout);

  long BN = (long)in_sizes[0] / D_EMB;     // 32768 tokens
  int nblocks = (int)(BN / T_TOK);         // 4096 blocks
  attn_main<<<nblocks, 256, 0, stream>>>(x, lnw, lnb, wqkv, wout, out);
}
